// MultiBoxLoss_24343874634295
// MI455X (gfx1250) — compile-verified, matrix operand order checked by reading, not run on hardware
//
#include <hip/hip_runtime.h>
#include <hip/hip_bf16.h>

#define BB 32
#define AA 100000
#define CC 81
#define TILE 128
#define NBIN1 512
#define NBIN2 4096
#define EXP_STRIDE 72   // halves per row: 64 data + 8 pad (144B rows, keeps ds_load_b128 16B-aligned)

typedef __attribute__((ext_vector_type(16))) _Float16 v16h;
typedef __attribute__((ext_vector_type(8)))  float    v8f;
typedef __attribute__((ext_vector_type(4)))  unsigned v4u;
typedef __attribute__((ext_vector_type(8)))  int      v8i;
typedef __attribute__((ext_vector_type(4)))  int      v4i;

#if __has_builtin(__builtin_amdgcn_tensor_load_to_lds) && __has_builtin(__builtin_amdgcn_s_wait_tensorcnt)
#define USE_TDM 1
#endif

union AFrag { v16h v; uint4 q[2]; };

__global__ void mbox_zero_kernel(unsigned* __restrict__ ws, int n, float* __restrict__ out) {
    int i = blockIdx.x * blockDim.x + threadIdx.x;
    if (i < n) ws[i] = 0u;
    if (i == 0) *out = 0.0f;
}

__global__ __launch_bounds__(TILE) void mbox_main(
    const float* __restrict__ loc_p, const float* __restrict__ cls_p,
    const float* __restrict__ loc_t, const int* __restrict__ cls_t,
    float* __restrict__ negloss, unsigned* __restrict__ hcnt, float* __restrict__ hsum,
    int* __restrict__ a_np, float* __restrict__ a_loc, float* __restrict__ a_cls)
{
    __shared__ __align__(16) float    lds_x[TILE * CC];           // 41472 B
    __shared__ __align__(16) _Float16 lds_exp[TILE * EXP_STRIDE]; // 18432 B
    __shared__ float    lds_sum[TILE];                            //   512 B
    __shared__ unsigned s_hc[NBIN1];                              //  2048 B
    __shared__ float    s_hs[NBIN1];                              //  2048 B
    __shared__ int   s_np;
    __shared__ float s_loc, s_cls;

    const int t = threadIdx.x;
    const int b = blockIdx.y;
    const int abase = blockIdx.x * TILE;
    const int nValid = min(TILE, AA - abase);

    if (t == 0) { s_np = 0; s_loc = 0.0f; s_cls = 0.0f; }
    for (int i = t; i < NBIN1; i += TILE) { s_hc[i] = 0u; s_hs[i] = 0.0f; }

    // ---- Stage cls tile into LDS ----
    {
        const float* gx = cls_p + (size_t)b * AA * CC + (size_t)abase * CC; // 16B aligned
        const int nE  = nValid * CC;
#if defined(USE_TDM)
        // Tensor Data Mover: 1D tile of nE f32 elements, global -> LDS.
        // Wave 0 only (tensor ops ignore EXEC; all-waves would issue 4 DMAs).
        if (t < 32) {
            const unsigned  lds_base = (unsigned)(unsigned long long)(void*)lds_x; // addr[31:0] = LDS byte addr
            const unsigned long long ga = (unsigned long long)(const void*)gx;
            v4u g0;
            g0[0] = 1u;                                         // count=1 user descriptor
            g0[1] = lds_base;                                   // lds_addr
            g0[2] = (unsigned)(ga & 0xFFFFFFFFu);               // global_addr[31:0]
            g0[3] = (unsigned)((ga >> 32) & 0x01FFFFFFu)        // global_addr[56:32]
                    | (2u << 30);                               // type = 2 ("image")
            v8i g1;
            g1[0] = (int)(2u << 16);                            // wg_mask=0, data_size=2 (4B)
            g1[1] = (int)(((unsigned)nE & 0xFFFFu) << 16);      // tensor_dim0[15:0] @ bits 63:48
            g1[2] = (int)((((unsigned)nE >> 16) & 0xFFFFu)      // tensor_dim0[31:16]
                    | (1u << 16));                              // tensor_dim1[15:0] = 1
            g1[3] = (int)(((unsigned)nE & 0xFFFFu) << 16);      // tensor_dim1[31:16]=0, tile_dim0=nE
            g1[4] = (int)(1u | (1u << 16));                     // tile_dim1=1, tile_dim2=1
            g1[5] = (int)nE;                                    // tensor_dim0_stride[31:0]
            g1[6] = 0;                                          // stride0[47:32]=0, stride1[15:0]=0
            g1[7] = 0;
            v4i g2 = {0, 0, 0, 0};
            v4i g3 = {0, 0, 0, 0};
#if defined(__clang_major__) && (__clang_major__ >= 23)
            v8i g4 = {0, 0, 0, 0, 0, 0, 0, 0};
            __builtin_amdgcn_tensor_load_to_lds(g0, g1, g2, g3, g4, 0);
#else
            __builtin_amdgcn_tensor_load_to_lds(g0, g1, g2, g3, 0);
#endif
            __builtin_amdgcn_s_wait_tensorcnt(0);
        }
#else
        // Fallback: coalesced + vectorized copy (global_load_b128 -> ds_store_b128)
        const int nE4 = nE >> 2;
        const float4* gx4 = (const float4*)gx;
        float4* lx4 = (float4*)lds_x;
        for (int i = t; i < nE4; i += TILE) lx4[i] = gx4[i];
        for (int i = (nE4 << 2) + t; i < nE; i += TILE) lds_x[i] = gx[i];
#endif
    }
    __syncthreads();

    // ---- Pass 1: per-anchor max (conflict-free: row stride 81 floats, 81 mod 64 = 17) ----
    const bool valid = (t < nValid);
    const int a = abase + t;
    int   ct = 0;
    float x0 = 0.0f, xt = 0.0f, m = -3.4e38f, s_tail = 0.0f;
    if (valid) {
        ct = cls_t[(size_t)b * AA + a];
        ct = min(max(ct, 0), CC - 1);
        const float* row = lds_x + t * CC;
        x0 = row[0];
        xt = row[ct];
        #pragma unroll 8
        for (int c = 0; c < CC; ++c) m = fmaxf(m, row[c]);
    }

    // ---- Pass 2: exp(x - max) -> f16 buffer for classes 0..63; tail 64..80 scalar ----
    {
        _Float16* er = lds_exp + t * EXP_STRIDE;
        if (valid) {
            const float* row = lds_x + t * CC;
            #pragma unroll 8
            for (int c = 0; c < 64; ++c) er[c] = (_Float16)__expf(row[c] - m);
            for (int c = 64; c < CC; ++c) s_tail += __expf(row[c] - m);
        } else {
            #pragma unroll 8
            for (int c = 0; c < 64; ++c) er[c] = (_Float16)0.0f;
        }
    }
    __syncthreads();

    // ---- WMMA: row-sum of exp over classes 0..63 via ones-matrix multiply ----
    // A = 16 anchors x 32 classes (f16), B = ones(32x16), C accumulates f32.
    {
        const v16h ones = { (_Float16)1.f,(_Float16)1.f,(_Float16)1.f,(_Float16)1.f,
                            (_Float16)1.f,(_Float16)1.f,(_Float16)1.f,(_Float16)1.f,
                            (_Float16)1.f,(_Float16)1.f,(_Float16)1.f,(_Float16)1.f,
                            (_Float16)1.f,(_Float16)1.f,(_Float16)1.f,(_Float16)1.f };
        const int wave = t >> 5;
        const int lane = t & 31;
        const int offB = (lane & 16) ? 16 : 0;        // lanes 16-31 hold K=8..15 / 24..31
        const char* ebase = (const char*)lds_exp;
        for (int g = wave; g < TILE / 16; g += 4) {    // 8 groups of 16 anchors, 4 waves
            const int anchor = g * 16 + (lane & 15);   // A-matrix M = lane%16
            v8f cacc = {};
            #pragma unroll
            for (int q = 0; q < 2; ++q) {              // class chunks 0..31, 32..63
                AFrag af;
                const char* p = ebase + anchor * (EXP_STRIDE * 2) + q * 64 + offB;
                af.q[0] = *(const uint4*)p;            // K = q*32 + off + 0..7
                af.q[1] = *(const uint4*)(p + 32);     // K = q*32 + off + 16..23
                cacc = __builtin_amdgcn_wmma_f32_16x16x32_f16(
                    false, af.v, false, ones, (short)0, cacc, false, false);
            }
            // D[m][0]: lane0 vgpr r -> M=r ; lane16 vgpr r -> M=8+r
            if (lane == 0) {
                lds_sum[g*16 + 0] = cacc[0]; lds_sum[g*16 + 1] = cacc[1];
                lds_sum[g*16 + 2] = cacc[2]; lds_sum[g*16 + 3] = cacc[3];
                lds_sum[g*16 + 4] = cacc[4]; lds_sum[g*16 + 5] = cacc[5];
                lds_sum[g*16 + 6] = cacc[6]; lds_sum[g*16 + 7] = cacc[7];
            } else if (lane == 16) {
                lds_sum[g*16 + 8]  = cacc[0]; lds_sum[g*16 + 9]  = cacc[1];
                lds_sum[g*16 + 10] = cacc[2]; lds_sum[g*16 + 11] = cacc[3];
                lds_sum[g*16 + 12] = cacc[4]; lds_sum[g*16 + 13] = cacc[5];
                lds_sum[g*16 + 14] = cacc[6]; lds_sum[g*16 + 15] = cacc[7];
            }
        }
    }
    __syncthreads();

    // ---- Per-anchor losses ----
    if (valid) {
        const float ssum = lds_sum[t] + s_tail;
        const float lse  = m + __logf(ssum);           // logsumexp
        const bool  pos  = (ct != 0);
        float nl = -1.0f;                              // sentinel: excluded from neg mining
        if (pos) {
            atomicAdd(&s_np, 1);
            atomicAdd(&s_cls, lse - xt);               // cross-entropy at target class
            const float4 lp = ((const float4*)loc_p)[(size_t)b * AA + a];
            const float4 lt = ((const float4*)loc_t)[(size_t)b * AA + a];
            float acc = 0.0f, d, ad;
            d = lp.x - lt.x; ad = fabsf(d); acc += (ad < 1.0f) ? 0.5f*d*d : ad - 0.5f;
            d = lp.y - lt.y; ad = fabsf(d); acc += (ad < 1.0f) ? 0.5f*d*d : ad - 0.5f;
            d = lp.z - lt.z; ad = fabsf(d); acc += (ad < 1.0f) ? 0.5f*d*d : ad - 0.5f;
            d = lp.w - lt.w; ad = fabsf(d); acc += (ad < 1.0f) ? 0.5f*d*d : ad - 0.5f;
            atomicAdd(&s_loc, acc);
        } else {
            nl = lse - x0;                             // -log_softmax[..,0], >= 0
            unsigned key = __float_as_uint(nl);
            unsigned bin = key >> 22;                  // monotone key for nonneg floats
            if (bin >= NBIN1) bin = NBIN1 - 1;
            atomicAdd(&s_hc[bin], 1u);
            atomicAdd(&s_hs[bin], nl);
        }
        negloss[(size_t)b * AA + a] = nl;
    }
    __syncthreads();

    // ---- Flush block-local accumulators / histogram ----
    if (t == 0) {
        if (s_np) atomicAdd(&a_np[b], s_np);
        atomicAdd(&a_loc[b], s_loc);
        atomicAdd(&a_cls[b], s_cls);
    }
    for (int i = t; i < NBIN1; i += TILE) {
        if (s_hc[i]) {
            atomicAdd(&hcnt[b * NBIN1 + i], s_hc[i]);
            atomicAdd(&hsum[b * NBIN1 + i], s_hs[i]);
        }
    }
}

__global__ __launch_bounds__(256) void mbox_select(
    const float* __restrict__ negloss, const unsigned* __restrict__ hcnt,
    const float* __restrict__ hsum, const int* __restrict__ a_np,
    const float* __restrict__ a_loc, const float* __restrict__ a_cls,
    float* __restrict__ out)
{
    __shared__ unsigned c1[NBIN1];
    __shared__ float    s1[NBIN1];
    __shared__ unsigned c2[NBIN2];
    __shared__ float    s2[NBIN2];
    __shared__ int   sT, sRem;
    __shared__ float sAbove;

    const int b = blockIdx.x;
    const int t = threadIdx.x;
    const int np = a_np[b];
    if (np == 0) return;                               // sample skipped (uniform exit)

    for (int i = t; i < NBIN1; i += 256) { c1[i] = hcnt[b*NBIN1+i]; s1[i] = hsum[b*NBIN1+i]; }
    for (int i = t; i < NBIN2; i += 256) { c2[i] = 0u; s2[i] = 0.0f; }
    __syncthreads();

    const long long k3 = (long long)3 * np;
    const long long ka = (long long)(AA - np);
    const long long k  = (k3 < ka) ? k3 : ka;

    if (t == 0) {
        long long cnt = 0; float sum = 0.0f; int T = 0; long long rem = k;
        for (int i = NBIN1 - 1; i >= 0; --i) {
            if (cnt + (long long)c1[i] >= k) { T = i; rem = k - cnt; break; }
            cnt += c1[i]; sum += s1[i];
        }
        sT = T; sRem = (int)rem; sAbove = sum;
    }
    __syncthreads();

    // ---- Refine inside threshold bin: sub-histogram on next 12 key bits ----
    const unsigned T = (unsigned)sT;
    for (int i = t; i < AA; i += 256) {
        const float v = negloss[(size_t)b * AA + i];
        const unsigned key = __float_as_uint(v);
        if ((key >> 22) == T && v >= 0.0f) {
            const unsigned sb = (key >> 10) & 0xFFFu;
            atomicAdd(&c2[sb], 1u);
            atomicAdd(&s2[sb], v);
        }
    }
    __syncthreads();

    if (t == 0) {
        const long long rem = sRem;
        long long cnt = 0; float sum = 0.0f; float partial = 0.0f;
        for (int i = NBIN2 - 1; i >= 0; --i) {
            if (cnt + (long long)c2[i] >= rem) {
                const long long take = rem - cnt;
                partial = sum + ((c2[i] > 0) ? (float)take * (s2[i] / (float)c2[i]) : 0.0f);
                break;
            }
            cnt += c2[i]; sum += s2[i];
        }
        const float cls_neg = sAbove + partial;
        const float denom = (float)np;
        const float total = (a_loc[b] + a_cls[b] + cls_neg) / denom;
        atomicAdd(out, total);
    }
}

extern "C" void kernel_launch(void* const* d_in, const int* in_sizes, int n_in,
                              void* d_out, int out_size, void* d_ws, size_t ws_size,
                              hipStream_t stream) {
    const float* loc_p = (const float*)d_in[0];
    const float* cls_p = (const float*)d_in[1];
    const float* loc_t = (const float*)d_in[2];
    const int*   cls_t = (const int*)d_in[3];
    float* out = (float*)d_out;

    char* w = (char*)d_ws;
    float*    negloss = (float*)w;
    size_t off = (size_t)BB * AA * sizeof(float);       // 12.8 MB
    unsigned* hcnt  = (unsigned*)(w + off); off += (size_t)BB * NBIN1 * 4;
    float*    hsum  = (float*)(w + off);    off += (size_t)BB * NBIN1 * 4;
    int*      a_np  = (int*)(w + off);      off += (size_t)BB * 4;
    float*    a_loc = (float*)(w + off);    off += (size_t)BB * 4;
    float*    a_cls = (float*)(w + off);    off += (size_t)BB * 4;
    (void)ws_size; (void)in_sizes; (void)n_in; (void)out_size;

    // Zero histograms + accumulators + output (contiguous region starting at hcnt)
    const int zwords = BB * NBIN1 * 2 + BB * 3;
    mbox_zero_kernel<<<(zwords + 255) / 256, 256, 0, stream>>>(hcnt, zwords, out);

    dim3 grid((AA + TILE - 1) / TILE, BB);
    mbox_main<<<grid, TILE, 0, stream>>>(loc_p, cls_p, loc_t, cls_t,
                                         negloss, hcnt, hsum, a_np, a_loc, a_cls);

    mbox_select<<<BB, 256, 0, stream>>>(negloss, hcnt, hsum, a_np, a_loc, a_cls, out);
}
